// SimpleMamba2Block_11690900980065
// MI455X (gfx1250) — matrix operand order repaired and checked by reading, not compiled, not run on hardware
//
#include <hip/hip_runtime.h>
#include <hip/hip_bf16.h>
#include <math.h>

// ---------------- problem constants ----------------
#define D_MODEL   1024
#define STATE     64
#define HEADS     8
#define INNER     2048           // EXPAND * D_MODEL
#define HEAD_DIM  256            // INNER / HEADS
#define PROJ_OUT  4232           // 2*INNER + 2*STATE + HEADS
#define NPAD1     4288           // PROJ_OUT padded up to 67*64 (WMMA tiling)
#define BATCH     2
#define SEQ       1024
#define ROWS      (BATCH * SEQ)  // 2048
#define EPS       1e-5f
#define QC        64             // SSD chunk length
#define NCHUNK    (SEQ / QC)     // 16 serial chunk iterations

typedef __bf16 bf16;
typedef __attribute__((ext_vector_type(16))) __bf16 v16bf;
typedef __attribute__((ext_vector_type(8)))  __bf16 v8bf;
typedef __attribute__((ext_vector_type(8)))  float  v8f;

#define WMMA_BF16(a, b, c) __builtin_amdgcn_wmma_f32_16x16x32_bf16( \
    false, (a), false, (b), (short)0, (c), false, false)

// ---- fragment loaders (contiguous k order, same for A and B) ----
// A operand (16x32 bf16): row = M, K = (lane>>4)*16 + 0..15 contiguous.
// Source: row-major, K contiguous.
__device__ __forceinline__ v16bf frag_a(const bf16* base, int row, int ld,
                                        int k0, int lane) {
    const bf16* p = base + (size_t)row * ld + k0 + ((lane >> 4) << 4);
    v8bf lo = *(const v8bf*)p;
    v8bf hi = *(const v8bf*)(p + 8);
    return __builtin_shufflevector(lo, hi, 0,1,2,3,4,5,6,7,8,9,10,11,12,13,14,15);
}
// B operand (32x16 bf16): lane = column N, K = (lane>>4)*16 + 0..15 contiguous.
// Source: N-major with K contiguous (i.e. B^T row-major).
__device__ __forceinline__ v16bf frag_b(const bf16* base, int col, int ld,
                                        int k0, int lane) {
    const bf16* p = base + (size_t)col * ld + k0 + ((lane >> 4) << 4);
    v8bf lo = *(const v8bf*)p;
    v8bf hi = *(const v8bf*)(p + 8);
    return __builtin_shufflevector(lo, hi, 0,1,2,3,4,5,6,7,8,9,10,11,12,13,14,15);
}
// B operand built from f32 LDS (state H), converted to bf16 on the fly.
__device__ __forceinline__ v16bf frag_b_f32(const float* base, int col, int ld,
                                            int k0, int lane) {
    const float* p = base + (size_t)col * ld + k0 + ((lane >> 4) << 4);
    v16bf r;
#pragma unroll
    for (int e = 0; e < 16; ++e) r[e] = (bf16)p[e];
    return r;
}

// ---------------- weight repack: W (K x N, f32, row-major) -> WT (NPAD x K, bf16) ----
__global__ __launch_bounds__(256)
void repack_wt_kernel(const float* __restrict__ W, bf16* __restrict__ WT,
                      int K, int N, int NPADv) {
    size_t i = (size_t)blockIdx.x * 256 + threadIdx.x;
    size_t total = (size_t)NPADv * K;
    if (i >= total) return;
    int n = (int)(i / K);
    int k = (int)(i % K);
    float v = (n < N) ? W[(size_t)k * N + n] : 0.0f;
    WT[i] = (bf16)v;
}

// ---------------- layernorm: one block per row, D_MODEL = 1024, 256 threads --------
__global__ __launch_bounds__(256)
void layernorm_bf16_kernel(const float* __restrict__ x,
                           const float* __restrict__ gamma,
                           const float* __restrict__ beta,
                           bf16* __restrict__ out) {
    const int row = blockIdx.x;
    const int tid = threadIdx.x;
    const float* xr = x + (size_t)row * D_MODEL;
    __shared__ float red[256];

    float v[4];
    float s = 0.0f;
#pragma unroll
    for (int i = 0; i < 4; ++i) { v[i] = xr[tid + i * 256]; s += v[i]; }
    red[tid] = s;
    __syncthreads();
    for (int off = 128; off > 0; off >>= 1) {
        if (tid < off) red[tid] += red[tid + off];
        __syncthreads();
    }
    const float mu = red[0] * (1.0f / D_MODEL);
    __syncthreads();

    float sv = 0.0f;
#pragma unroll
    for (int i = 0; i < 4; ++i) { float d = v[i] - mu; sv += d * d; }
    red[tid] = sv;
    __syncthreads();
    for (int off = 128; off > 0; off >>= 1) {
        if (tid < off) red[tid] += red[tid + off];
        __syncthreads();
    }
    const float rs = rsqrtf(red[0] * (1.0f / D_MODEL) + EPS);

    bf16* orow = out + (size_t)row * D_MODEL;
#pragma unroll
    for (int i = 0; i < 4; ++i) {
        int c = tid + i * 256;
        orow[c] = (bf16)((v[i] - mu) * rs * gamma[c] + beta[c]);
    }
}

// ---------------- WMMA GEMM: C[M x ldc] = A[M x K] * BT[N x K]^T (+bias)(+resid) ---
__global__ __launch_bounds__(128)
void wmma_gemm_bf16_kernel(const bf16* __restrict__ A,
                           const bf16* __restrict__ BT,
                           float* __restrict__ C,
                           const float* __restrict__ bias,   // may be null
                           const float* __restrict__ resid,  // may be null, ld=Nreal
                           int M, int K, int ldc, int Nreal) {
    const int lane = threadIdx.x & 31;
    const int wave = threadIdx.x >> 5;
    const int n0   = blockIdx.x * 64;
    const int m0   = blockIdx.y * 256 + wave * 64;

    v8f acc[4][4];
#pragma unroll
    for (int mi = 0; mi < 4; ++mi)
#pragma unroll
        for (int ni = 0; ni < 4; ++ni) acc[mi][ni] = (v8f){0,0,0,0,0,0,0,0};

    for (int k0 = 0; k0 < K; k0 += 32) {
        v16bf afrag[4], bfrag[4];
#pragma unroll
        for (int mi = 0; mi < 4; ++mi)
            afrag[mi] = frag_a(A, m0 + mi * 16 + (lane & 15), K, k0, lane);
#pragma unroll
        for (int ni = 0; ni < 4; ++ni)
            bfrag[ni] = frag_b(BT, n0 + ni * 16 + (lane & 15), K, k0, lane);
#pragma unroll
        for (int mi = 0; mi < 4; ++mi)
#pragma unroll
            for (int ni = 0; ni < 4; ++ni)
                acc[mi][ni] = WMMA_BF16(afrag[mi], bfrag[ni], acc[mi][ni]);
    }

    const int ccol = lane & 15;
    const int crow = (lane >> 4) * 8;
#pragma unroll
    for (int mi = 0; mi < 4; ++mi) {
#pragma unroll
        for (int ni = 0; ni < 4; ++ni) {
            const int col   = n0 + ni * 16 + ccol;
            const int rbase = m0 + mi * 16 + crow;
            float badd = (bias && col < Nreal) ? bias[col] : 0.0f;
#pragma unroll
            for (int r = 0; r < 8; ++r) {
                float v = acc[mi][ni][r] + badd;
                if (resid && col < Nreal)
                    v += resid[(size_t)(rbase + r) * Nreal + col];
                C[(size_t)(rbase + r) * ldc + col] = v;
            }
        }
    }
}

// ---------------- chunked SSD scan (Mamba2), all matmuls on WMMA -------------------
// grid = BATCH*HEADS blocks, 256 threads (8 waves). Serial only over 16 chunks.
// LDS: H (f32 256x64), X (bf16 [p][s]), C/Cs (bf16 [t][n]), B ([s][n]), Bw ([n][s]),
//      G ([t][s]), cumulative decay + dt.
#define SMEM_H    0
#define SMEM_X    (SMEM_H + HEAD_DIM * STATE * 4)            // 65536
#define SMEM_CR   (SMEM_X + HEAD_DIM * QC * 2)               // +32768
#define SMEM_CS   (SMEM_CR + QC * STATE * 2)
#define SMEM_BR   (SMEM_CS + QC * STATE * 2)
#define SMEM_BW   (SMEM_BR + QC * STATE * 2)
#define SMEM_G    (SMEM_BW + QC * STATE * 2)
#define SMEM_ST   (SMEM_G + QC * QC * 2)
#define SMEM_DT   (SMEM_ST + QC * 4)
#define SMEM_TOTAL (SMEM_DT + QC * 4)                        // ~139.8 KB

__global__ __launch_bounds__(256)
void mamba_chunk_kernel(const float* __restrict__ zx,     // ROWS x NPAD1
                        const float* __restrict__ A_log,
                        const float* __restrict__ D_skip,
                        const float* __restrict__ dt_bias,
                        bf16* __restrict__ y) {            // ROWS x INNER
    extern __shared__ char smem[];
    float* sH  = (float*)(smem + SMEM_H);
    bf16*  sX  = (bf16*)(smem + SMEM_X);
    bf16*  sCr = (bf16*)(smem + SMEM_CR);
    bf16*  sCs = (bf16*)(smem + SMEM_CS);
    bf16*  sBr = (bf16*)(smem + SMEM_BR);
    bf16*  sBw = (bf16*)(smem + SMEM_BW);
    bf16*  sG  = (bf16*)(smem + SMEM_G);
    float* sSt = (float*)(smem + SMEM_ST);
    float* sDt = (float*)(smem + SMEM_DT);

    const int tid  = threadIdx.x;
    const int lane = tid & 31;
    const int wave = tid >> 5;
    const int h    = blockIdx.x & 7;
    const int b    = blockIdx.x >> 3;

    const float Ah  = -__expf(A_log[h]);
    const float Dh  = D_skip[h];
    const float dtb = dt_bias[h];

    for (int i = tid; i < HEAD_DIM * STATE; i += 256) sH[i] = 0.0f;
    __syncthreads();

    for (int c = 0; c < NCHUNK; ++c) {
        const int row0 = b * SEQ + c * QC;

        // ---- phase 0a: dt = softplus, cumulative log-decay s_t ----
        if (tid < QC) {
            float raw = zx[(size_t)(row0 + tid) * NPAD1 + 2 * INNER + 2 * STATE + h] + dtb;
            float dtv = (raw > 20.0f) ? raw : log1pf(__expf(raw));
            sDt[tid] = dtv;
            sSt[tid] = dtv * Ah;
        }
        __syncthreads();
        if (tid == 0) {
            float run = 0.0f;
            for (int t = 0; t < QC; ++t) { run += sSt[t]; sSt[t] = run; }
        }
        __syncthreads();
        const float sLast = sSt[QC - 1];

        // ---- phase 0b: stage B, Bw, C, Cs ----
        for (int i = tid; i < QC * STATE; i += 256) {
            int t = i >> 6, n = i & 63;
            size_t rb = (size_t)(row0 + t) * NPAD1;
            float bv = zx[rb + 2 * INNER + n];
            float cv = zx[rb + 2 * INNER + STATE + n];
            float st = sSt[t];
            sBr[i]           = (bf16)bv;                                  // [s][n]
            sBw[n * QC + t]  = (bf16)(bv * sDt[t] * __expf(sLast - st));  // [n][s]
            sCr[i]           = (bf16)cv;                                  // [t][n]
            sCs[i]           = (bf16)(cv * __expf(st));                   // [t][n]
        }
        // ---- phase 0c: stage X transposed [p][s] (coalesced global reads) ----
        for (int i = tid; i < HEAD_DIM * QC; i += 256) {
            int p = i & 255, t = i >> 8;
            float xv = zx[(size_t)(row0 + t) * NPAD1 + INNER + h * HEAD_DIM + p];
            sX[p * QC + t] = (bf16)xv;
        }
        __syncthreads();

        // ---- phase 1: G0 = C B^T (t x s), mask * exp(s_t - s_s) * dt_s -> sG ----
        {
#pragma unroll
            for (int j = 0; j < 2; ++j) {
                int idx = wave * 2 + j;
                int mi = idx >> 2, ni = idx & 3;
                v8f acc = (v8f){0,0,0,0,0,0,0,0};
#pragma unroll
                for (int k0 = 0; k0 < STATE; k0 += 32) {
                    v16bf a  = frag_a(sCr, mi * 16 + (lane & 15), STATE, k0, lane);
                    v16bf bb = frag_b(sBr, ni * 16 + (lane & 15), STATE, k0, lane);
                    acc = WMMA_BF16(a, bb, acc);
                }
                int s  = ni * 16 + (lane & 15);
                int tb = mi * 16 + ((lane >> 4) * 8);
                float ds = sDt[s], ss = sSt[s];
#pragma unroll
                for (int r = 0; r < 8; ++r) {
                    int t = tb + r;
                    float v = (s <= t) ? acc[r] * ds * __expf(sSt[t] - ss) : 0.0f;
                    sG[t * QC + s] = (bf16)v;
                }
            }
        }
        __syncthreads();

        // ---- phase 2: Y = G @ X + Cs @ H^T ; fuse D-skip + SiLU gate -> y ----
        {
            int mi = wave & 3;      // t tile
            int ph = wave >> 2;     // p half
            v8f acc[8];
#pragma unroll
            for (int ni = 0; ni < 8; ++ni) acc[ni] = (v8f){0,0,0,0,0,0,0,0};
#pragma unroll
            for (int k0 = 0; k0 < QC; k0 += 32) {
                v16bf a = frag_a(sG, mi * 16 + (lane & 15), QC, k0, lane);
#pragma unroll
                for (int ni = 0; ni < 8; ++ni) {
                    int p = ph * 128 + ni * 16 + (lane & 15);
                    acc[ni] = WMMA_BF16(a, frag_b(sX, p, QC, k0, lane), acc[ni]);
                }
            }
#pragma unroll
            for (int k0 = 0; k0 < STATE; k0 += 32) {
                v16bf a = frag_a(sCs, mi * 16 + (lane & 15), STATE, k0, lane);
#pragma unroll
                for (int ni = 0; ni < 8; ++ni) {
                    int p = ph * 128 + ni * 16 + (lane & 15);
                    acc[ni] = WMMA_BF16(a, frag_b_f32(sH, p, STATE, k0, lane), acc[ni]);
                }
            }
            int tb = mi * 16 + ((lane >> 4) * 8);
#pragma unroll
            for (int ni = 0; ni < 8; ++ni) {
                int p = ph * 128 + ni * 16 + (lane & 15);
#pragma unroll
                for (int r = 0; r < 8; ++r) {
                    int t = tb + r;
                    float xv = (float)sX[p * QC + t];
                    float zv = zx[(size_t)(row0 + t) * NPAD1 + h * HEAD_DIM + p];
                    float yv = (acc[ni][r] + Dh * xv) * (zv / (1.0f + __expf(-zv)));
                    y[(size_t)(row0 + t) * INNER + h * HEAD_DIM + p] = (bf16)yv;
                }
            }
        }
        __syncthreads();

        // ---- phase 3: dH = X^T @ Bw ; H = exp(s_last)*H + dH ----
        {
            float dAll = __expf(sLast);
#pragma unroll
            for (int j = 0; j < 2; ++j) {
                int mi = wave * 2 + j;   // p tile 0..15
                v8f acc[4];
#pragma unroll
                for (int ni = 0; ni < 4; ++ni) acc[ni] = (v8f){0,0,0,0,0,0,0,0};
#pragma unroll
                for (int k0 = 0; k0 < QC; k0 += 32) {
                    v16bf a = frag_a(sX, mi * 16 + (lane & 15), QC, k0, lane);
#pragma unroll
                    for (int ni = 0; ni < 4; ++ni)
                        acc[ni] = WMMA_BF16(a,
                            frag_b(sBw, ni * 16 + (lane & 15), QC, k0, lane), acc[ni]);
                }
                int pb = mi * 16 + ((lane >> 4) * 8);
#pragma unroll
                for (int ni = 0; ni < 4; ++ni) {
                    int n = ni * 16 + (lane & 15);
#pragma unroll
                    for (int r = 0; r < 8; ++r) {
                        int p = pb + r;
                        sH[p * STATE + n] = dAll * sH[p * STATE + n] + acc[ni][r];
                    }
                }
            }
        }
        __syncthreads();
    }
}

// ---------------- host glue --------------------------------------------------------
static inline size_t align256(size_t x) { return (x + 255) & ~(size_t)255; }

extern "C" void kernel_launch(void* const* d_in, const int* in_sizes, int n_in,
                              void* d_out, int out_size, void* d_ws, size_t ws_size,
                              hipStream_t stream) {
    const float* input    = (const float*)d_in[0];
    const float* ln_gamma = (const float*)d_in[1];
    const float* ln_beta  = (const float*)d_in[2];
    const float* W_in     = (const float*)d_in[3];
    const float* b_in     = (const float*)d_in[4];
    const float* A_log    = (const float*)d_in[5];
    const float* D_skip   = (const float*)d_in[6];
    const float* dt_bias  = (const float*)d_in[7];
    const float* W_out    = (const float*)d_in[8];
    const float* b_out    = (const float*)d_in[9];

    char* ws = (char*)d_ws;
    bf16*  A1    = (bf16*)ws;  ws += align256((size_t)ROWS * D_MODEL * sizeof(bf16));
    bf16*  WinT  = (bf16*)ws;  ws += align256((size_t)NPAD1 * D_MODEL * sizeof(bf16));
    float* ZX    = (float*)ws; ws += align256((size_t)ROWS * NPAD1 * sizeof(float));
    bf16*  Y2    = (bf16*)ws;  ws += align256((size_t)ROWS * INNER * sizeof(bf16));
    bf16*  WoutT = (bf16*)ws;  ws += align256((size_t)D_MODEL * INNER * sizeof(bf16));

    {
        size_t tot = (size_t)NPAD1 * D_MODEL;
        repack_wt_kernel<<<dim3((unsigned)((tot + 255) / 256)), dim3(256), 0, stream>>>(
            W_in, WinT, D_MODEL, PROJ_OUT, NPAD1);
    }
    {
        size_t tot = (size_t)D_MODEL * INNER;
        repack_wt_kernel<<<dim3((unsigned)((tot + 255) / 256)), dim3(256), 0, stream>>>(
            W_out, WoutT, INNER, D_MODEL, D_MODEL);
    }

    layernorm_bf16_kernel<<<dim3(ROWS), dim3(256), 0, stream>>>(
        input, ln_gamma, ln_beta, A1);

    // input projection: ZX = A1 (2048x1024) @ W_in (1024x4232) + b_in
    wmma_gemm_bf16_kernel<<<dim3(NPAD1 / 64, ROWS / 256), dim3(128), 0, stream>>>(
        A1, WinT, ZX, b_in, nullptr, ROWS, D_MODEL, NPAD1, PROJ_OUT);

    // chunked SSD scan + gating (WMMA), serial depth = 16 chunks
    mamba_chunk_kernel<<<dim3(BATCH * HEADS), dim3(256), SMEM_TOTAL, stream>>>(
        ZX, A_log, D_skip, dt_bias, Y2);

    // output projection + bias + residual -> d_out (f32)
    wmma_gemm_bf16_kernel<<<dim3(D_MODEL / 64, ROWS / 256), dim3(128), 0, stream>>>(
        Y2, WoutT, (float*)d_out, b_out, input, ROWS, INNER, D_MODEL, D_MODEL);
}
